// CrossMamba_52209622450779
// MI455X (gfx1250) — matrix operand order repaired and checked by reading, not compiled, not run on hardware
//
#include <hip/hip_runtime.h>
#include <hip/hip_bf16.h>

// ---------------------------------------------------------------------------
// CDNA5 / gfx1250 wave32 WMMA implementation of the CrossMamba block.
// All tensor contractions lower to v_wmma_f32_16x16x32_bf16.
// GEMMs are 2x2 register-blocked per wave with double-buffered fragments.
// ---------------------------------------------------------------------------

typedef __attribute__((ext_vector_type(16))) __bf16 v16bf;
typedef __attribute__((ext_vector_type(8)))  __bf16 bf16x8;
typedef __attribute__((ext_vector_type(8)))  float  v8f;

#define L_TOK   1024
#define DMODEL  384
#define DSSM    512
#define NSTATE  512      // 4 directions * 128 concatenated (per reference reshape)
#define NH      32       // K * NHEADS
#define HD      64
#define DMLP    256
#define CHUNKL  256
#define NCHUNK  4
#define BCDT_C  264
#define BCDT_P  288      // padded to multiple of 32 for super-tiled WMMA GEMM
#define DCAT    768

static __device__ __forceinline__ float sigmoidf_(float x){ return 1.f/(1.f+__expf(-x)); }
static __device__ __forceinline__ float siluf_(float x){ return x*sigmoidf_(x); }
static __device__ __forceinline__ float softplusf_(float x){ return (x>20.f)? x : log1pf(__expf(x)); }

static __device__ __forceinline__ v8f wmma_bf16(v16bf a, v16bf b, v8f c) {
    // D = A(16x32) * B(32x16) + C, f32 accum
    return __builtin_amdgcn_wmma_f32_16x16x32_bf16(false, a, false, b, (short)0, c, false, false);
}

// A-fragment (16x32 bf16, M = lane%16): two contiguous 8-element runs at
// k = kb + 8*(lane/16) and k = kb + 16 + 8*(lane/16).
static __device__ __forceinline__ v16bf load_a_frag(const __bf16* A, int lda, int mrow, int kb, int lane) {
    const __bf16* base = A + (size_t)mrow*lda + kb + 8*(lane>>4);
    bf16x8 lo = *(const bf16x8*)(base);
    bf16x8 hi = *(const bf16x8*)(base + 16);
    v16bf a;
    #pragma unroll
    for (int i=0;i<8;i++){ a[i]=lo[i]; a[8+i]=hi[i]; }
    return a;
}

// B-fragment (32x16 bf16, N = lane%16) from Bt stored N-major (row n, contiguous k):
// two contiguous 8-element runs at k = kb + 8*(lane/16) and k = kb + 16 + 8*(lane/16).
static __device__ __forceinline__ v16bf load_b_frag(const __bf16* Bt, int ldb, int ncol, int kb, int lane) {
    const __bf16* base = Bt + (size_t)ncol*ldb + kb + 8*(lane>>4);
    bf16x8 b0 = *(const bf16x8*)(base);
    bf16x8 b1 = *(const bf16x8*)(base + 16);
    v16bf b;
    #pragma unroll
    for (int i=0;i<8;i++){ b[i]=b0[i]; b[8+i]=b1[i]; }
    return b;
}

// ---------------------------------------------------------------------------
// Generic WMMA GEMM: C[M,N] = A[M,K] * Bt[N,K]^T   (bf16 in, f32 out)
// 4 waves per block; each wave owns a 32x32 super-tile (2x2 WMMA tiles) with
// software-pipelined (double-buffered) fragment loads.
// Requires M%32==0, N%32==0, K%32==0.
// ---------------------------------------------------------------------------
__global__ void gemm_bf16_kernel(const __bf16* __restrict__ A, const __bf16* __restrict__ Bt,
                                 float* __restrict__ C,
                                 int M, int N, int Kd, int lda, int ldb, int ldc)
{
    int lane = threadIdx.x & 31;
    int wave = threadIdx.x >> 5;
    int stn  = N >> 5;
    int st   = (M >> 5) * stn;
    int s    = blockIdx.x * (blockDim.x >> 5) + wave;
    if (s >= st) return;
    int m0 = (s / stn) * 32;
    int n0 = (s % stn) * 32;
    int mrow = m0 + (lane & 15);
    int ncol = n0 + (lane & 15);

    v8f acc00 = {}, acc01 = {}, acc10 = {}, acc11 = {};
    v16bf a0 = load_a_frag(A, lda, mrow,      0, lane);
    v16bf a1 = load_a_frag(A, lda, mrow + 16, 0, lane);
    v16bf b0 = load_b_frag(Bt, ldb, ncol,     0, lane);
    v16bf b1 = load_b_frag(Bt, ldb, ncol + 16,0, lane);

    for (int kb = 0; kb < Kd; kb += 32) {
        int kn = kb + 32;
        v16bf na0 = a0, na1 = a1, nb0 = b0, nb1 = b1;
        if (kn < Kd) {
            __builtin_prefetch(A + (size_t)mrow*lda + kn + 32, 0, 0);
            __builtin_prefetch(Bt + (size_t)ncol*ldb + kn + 32, 0, 0);
            na0 = load_a_frag(A, lda, mrow,      kn, lane);
            na1 = load_a_frag(A, lda, mrow + 16, kn, lane);
            nb0 = load_b_frag(Bt, ldb, ncol,     kn, lane);
            nb1 = load_b_frag(Bt, ldb, ncol + 16,kn, lane);
        }
        acc00 = wmma_bf16(a0, b0, acc00);
        acc01 = wmma_bf16(a0, b1, acc01);
        acc10 = wmma_bf16(a1, b0, acc10);
        acc11 = wmma_bf16(a1, b1, acc11);
        a0 = na0; a1 = na1; b0 = nb0; b1 = nb1;
    }

    int mr = 8*(lane>>4);
    int nc = lane & 15;
    #pragma unroll
    for (int r=0;r<8;r++) {
        C[(size_t)(m0 + mr + r     )*ldc + n0 + nc     ] = acc00[r];
        C[(size_t)(m0 + mr + r     )*ldc + n0 + nc + 16] = acc01[r];
        C[(size_t)(m0 + mr + r + 16)*ldc + n0 + nc     ] = acc10[r];
        C[(size_t)(m0 + mr + r + 16)*ldc + n0 + nc + 16] = acc11[r];
    }
}

// ---------------------------------------------------------------------------
// fp32 -> bf16 copy / padded-transpose converters
// ---------------------------------------------------------------------------
__global__ void f32_to_bf16_kernel(const float* __restrict__ in, __bf16* __restrict__ out, int n) {
    int i = blockIdx.x*blockDim.x + threadIdx.x;
    if (i < n) out[i] = (__bf16)in[i];
}
// in: fp32 [K][N] row-major  ->  out: bf16 [Npad][K], zero-padded rows n>=N
__global__ void transpose_bf16_kernel(const float* __restrict__ in, __bf16* __restrict__ out,
                                      int K, int N, int Npad) {
    int i = blockIdx.x*blockDim.x + threadIdx.x;
    if (i >= Npad*K) return;
    int n = i / K, k = i % K;
    out[i] = (n < N) ? (__bf16)in[(size_t)k*N + n] : (__bf16)0.f;
}

// ---------------------------------------------------------------------------
// 3x3 depthwise conv (SAME, zero pad) + SiLU.
// in: fp32 [1024 tokens][ldin ch] (token-major, t=h*32+w)  out: fp32 [C][1024] ch-major
// ---------------------------------------------------------------------------
__global__ void dwconv_silu_kernel(const float* __restrict__ X, const float* __restrict__ Wc,
                                   const float* __restrict__ bias, float* __restrict__ out,
                                   int C, int ldin)
{
    int i = blockIdx.x*blockDim.x + threadIdx.x;
    if (i >= C*L_TOK) return;
    int c = i >> 10, t = i & 1023;
    int h = t >> 5, w = t & 31;
    float s = bias[c];
    #pragma unroll
    for (int dh=-1; dh<=1; dh++) {
        int hh = h + dh; if (hh < 0 || hh > 31) continue;
        #pragma unroll
        for (int dw=-1; dw<=1; dw++) {
            int ww = w + dw; if (ww < 0 || ww > 31) continue;
            s += X[(size_t)(hh*32+ww)*ldin + c] * Wc[c*9 + (dh+1)*3 + (dw+1)];
        }
    }
    out[(size_t)c*L_TOK + t] = siluf_(s);
}

// spatial location for direction k at scan position t
static __device__ __forceinline__ int dir_loc(int k, int t) {
    int tt = (k & 2) ? (1023 - t) : t;
    return (k & 1) ? ((tt & 31)*32 + (tt >> 5)) : tt;
}

// ---------------------------------------------------------------------------
// Build direction sequences: dt (softplus), xdt^T (bf16), Br/Cr (bf16) and Br^T.
// ---------------------------------------------------------------------------
__global__ void build_seq_kernel(const float* __restrict__ convXS, const float* __restrict__ convBC,
                                 const float* __restrict__ dt_bias,
                                 float* __restrict__ dtbuf,          // [1024][32]
                                 __bf16* __restrict__ xdtT,          // [32][64][1024]
                                 __bf16* __restrict__ CrB,           // [1024][512]
                                 __bf16* __restrict__ BrB,           // [1024][512]
                                 __bf16* __restrict__ BrT)           // [512][1024]
{
    int i = blockIdx.x*blockDim.x + threadIdx.x;
    if (i >= 4*L_TOK) return;
    int k = i >> 10, t = i & 1023;
    int loc = dir_loc(k, t);
    float dts[8];
    #pragma unroll
    for (int hh=0; hh<8; hh++) {
        int head = k*8 + hh;
        float dt = softplusf_(convBC[(size_t)(256+hh)*L_TOK + loc] + dt_bias[head]);
        dtbuf[(size_t)t*NH + head] = dt;
        dts[hh] = dt;
    }
    for (int hh=0; hh<8; hh++) {
        int head = k*8 + hh;
        for (int p=0; p<HD; p++) {
            float xv = convXS[(size_t)(hh*HD+p)*L_TOK + loc];
            xdtT[((size_t)head*HD + p)*L_TOK + t] = (__bf16)(xv * dts[hh]);
        }
    }
    for (int c=0; c<128; c++) {
        float bv = convBC[(size_t)c*L_TOK + loc];
        float cv = convBC[(size_t)(128+c)*L_TOK + loc];
        int ng = k*128 + c;
        BrB[(size_t)t*NSTATE + ng] = (__bf16)bv;
        CrB[(size_t)t*NSTATE + ng] = (__bf16)cv;
        BrT[(size_t)ng*L_TOK + t]  = (__bf16)bv;
    }
}

// dA cumsum per (chunk, head)
__global__ void cumsum_kernel(const float* __restrict__ dtbuf, const float* __restrict__ A_logs,
                              float* __restrict__ dAcs)
{
    int i = blockIdx.x*blockDim.x + threadIdx.x;
    if (i >= NCHUNK*NH) return;
    int z = i / NH, h = i % NH;
    float Ah = -__expf(A_logs[h]);
    float s = 0.f;
    for (int c=0; c<CHUNKL; c++) {
        int t = z*CHUNKL + c;
        s += dtbuf[(size_t)t*NH + h] * Ah;
        dAcs[(size_t)t*NH + h] = s;
    }
}

// ---------------------------------------------------------------------------
// Ydiag: per (chunk z, head h):  Y[i,p] = sum_{j<=i} CB[i,j]*exp(s_i - s_j) * xdt[j,p]
// Each wave owns 2 row-tiles (it=2w,2w+1) x 4 p-tiles; per j-block:
// 4 shared B-fragments + 2 computed A-fragments -> 8 WMMAs.
// Writes yacc = Ydiag + D*x.
// ---------------------------------------------------------------------------
__global__ void ydiag_kernel(const float* __restrict__ CB,     // [4][256][256]
                             const float* __restrict__ dAcs,   // [1024][32]
                             const __bf16* __restrict__ xdtT,  // [32][64][1024]
                             const float* __restrict__ convXS, // [512][1024]
                             const float* __restrict__ Ds,
                             float* __restrict__ yacc)         // [1024][32][64]
{
    int z = blockIdx.x >> 5, h = blockIdx.x & 31;
    __shared__ float sAc[CHUNKL];
    sAc[threadIdx.x] = dAcs[(size_t)(z*CHUNKL + threadIdx.x)*NH + h];
    __syncthreads();
    int lane = threadIdx.x & 31, wave = threadIdx.x >> 5;
    const float* CBz = CB + (size_t)z*CHUNKL*CHUNKL;
    const __bf16* xB = xdtT + (size_t)(h*HD)*L_TOK;
    float Dh = Ds[h];
    int kdir = h >> 3, hh = h & 7;

    int it0 = 2*wave, it1 = 2*wave + 1;
    int i0 = it0*16 + (lane & 15);
    int i1 = it1*16 + (lane & 15);
    float si0 = sAc[i0], si1 = sAc[i1];
    const float* crow0 = CBz + (size_t)i0*CHUNKL;
    const float* crow1 = CBz + (size_t)i1*CHUNKL;
    int half = lane >> 4;

    v8f acc0[4] = {}, acc1[4] = {};
    int jmax0 = it0*16 + 15;
    int jmax1 = it1*16 + 15;
    for (int jb = 0; jb <= jmax1; jb += 32) {
        v16bf b[4];
        #pragma unroll
        for (int pt=0; pt<4; pt++)
            b[pt] = load_b_frag(xB, L_TOK, pt*16 + (lane & 15), z*CHUNKL + jb, lane);
        int j0 = jb + 8*half;
        // upper row-tile (always contributes at jb <= jmax1)
        {
            v16bf a;
            #pragma unroll
            for (int e=0; e<8; e++) {
                int j1 = j0 + e, j2 = j0 + 16 + e;
                a[e]   = (__bf16)((j1 <= i1) ? crow1[j1]*__expf(si1 - sAc[j1]) : 0.f);
                a[8+e] = (__bf16)((j2 <= i1) ? crow1[j2]*__expf(si1 - sAc[j2]) : 0.f);
            }
            #pragma unroll
            for (int pt=0; pt<4; pt++) acc1[pt] = wmma_bf16(a, b[pt], acc1[pt]);
        }
        if (jb <= jmax0) {
            v16bf a;
            #pragma unroll
            for (int e=0; e<8; e++) {
                int j1 = j0 + e, j2 = j0 + 16 + e;
                a[e]   = (__bf16)((j1 <= i0) ? crow0[j1]*__expf(si0 - sAc[j1]) : 0.f);
                a[8+e] = (__bf16)((j2 <= i0) ? crow0[j2]*__expf(si0 - sAc[j2]) : 0.f);
            }
            #pragma unroll
            for (int pt=0; pt<4; pt++) acc0[pt] = wmma_bf16(a, b[pt], acc0[pt]);
        }
    }
    int mr = 8*half;
    #pragma unroll
    for (int pt=0; pt<4; pt++) {
        int p = pt*16 + (lane & 15);
        #pragma unroll
        for (int r=0; r<8; r++) {
            int t0 = z*CHUNKL + it0*16 + mr + r;
            int t1 = z*CHUNKL + it1*16 + mr + r;
            float xv0 = convXS[(size_t)(hh*HD + p)*L_TOK + dir_loc(kdir, t0)];
            float xv1 = convXS[(size_t)(hh*HD + p)*L_TOK + dir_loc(kdir, t1)];
            yacc[((size_t)t0*NH + h)*HD + p] = acc0[pt][r] + Dh*xv0;
            yacc[((size_t)t1*NH + h)*HD + p] = acc1[pt][r] + Dh*xv1;
        }
    }
}

// ---------------------------------------------------------------------------
// states: per (z,h): S[n,p] = sum_c Br[c,n]*exp(s_last - s_c)*xdt[c,p]
// M=512(n), K=256(c), N=64(p). Wave owns 4 n-tiles processed as 2 pairs,
// each pair: 4 shared B-fragments + 2 scaled A-fragments -> 8 WMMAs per c-block.
// Output layout [z][h][p][n] (n contiguous for Yoff B-frags).
// ---------------------------------------------------------------------------
__global__ void states_kernel(const __bf16* __restrict__ BrT,   // [512][1024]
                              const float* __restrict__ dAcs,
                              const __bf16* __restrict__ xdtT,
                              float* __restrict__ states)       // [4][32][64][512]
{
    int z = blockIdx.x >> 5, h = blockIdx.x & 31;
    __shared__ float sDec[CHUNKL];
    {
        float last = dAcs[(size_t)(z*CHUNKL + CHUNKL-1)*NH + h];
        sDec[threadIdx.x] = __expf(last - dAcs[(size_t)(z*CHUNKL + threadIdx.x)*NH + h]);
    }
    __syncthreads();
    int lane = threadIdx.x & 31, wave = threadIdx.x >> 5;
    int half = lane >> 4;
    const __bf16* xB = xdtT + (size_t)(h*HD)*L_TOK;

    for (int rep = 0; rep < 2; rep++) {
        int ntA = wave*4 + rep*2, ntB = ntA + 1;
        int nrowA = ntA*16 + (lane & 15);
        int nrowB = ntB*16 + (lane & 15);
        v8f accA[4] = {}, accB[4] = {};
        for (int cb = 0; cb < CHUNKL; cb += 32) {
            v16bf b[4];
            #pragma unroll
            for (int pt=0; pt<4; pt++)
                b[pt] = load_b_frag(xB, L_TOK, pt*16 + (lane & 15), z*CHUNKL + cb, lane);
            int c0 = cb + 8*half;
            v16bf aA, aB;
            {
                const __bf16* base = BrT + (size_t)nrowA*L_TOK + z*CHUNKL + c0;
                bf16x8 a0v = *(const bf16x8*)base;
                bf16x8 a1v = *(const bf16x8*)(base + 16);
                #pragma unroll
                for (int e=0; e<8; e++) {
                    aA[e]   = (__bf16)((float)a0v[e] * sDec[c0 + e]);
                    aA[8+e] = (__bf16)((float)a1v[e] * sDec[c0 + 16 + e]);
                }
            }
            {
                const __bf16* base = BrT + (size_t)nrowB*L_TOK + z*CHUNKL + c0;
                bf16x8 a0v = *(const bf16x8*)base;
                bf16x8 a1v = *(const bf16x8*)(base + 16);
                #pragma unroll
                for (int e=0; e<8; e++) {
                    aB[e]   = (__bf16)((float)a0v[e] * sDec[c0 + e]);
                    aB[8+e] = (__bf16)((float)a1v[e] * sDec[c0 + 16 + e]);
                }
            }
            #pragma unroll
            for (int pt=0; pt<4; pt++) {
                accA[pt] = wmma_bf16(aA, b[pt], accA[pt]);
                accB[pt] = wmma_bf16(aB, b[pt], accB[pt]);
            }
        }
        int nr = 8*half;
        #pragma unroll
        for (int pt=0; pt<4; pt++) {
            int p = pt*16 + (lane & 15);
            #pragma unroll
            for (int r=0; r<8; r++) {
                states[((size_t)(z*NH + h)*HD + p)*NSTATE + ntA*16 + nr + r] = accA[pt][r];
                states[((size_t)(z*NH + h)*HD + p)*NSTATE + ntB*16 + nr + r] = accB[pt][r];
            }
        }
    }
}

// inter-chunk recurrence: prev[z] = carry;  carry = carry*cdec[z] + states[z]
__global__ void recurrence_kernel(const float* __restrict__ states, const float* __restrict__ dAcs,
                                  __bf16* __restrict__ prevB)
{
    int i = blockIdx.x*blockDim.x + threadIdx.x;
    if (i >= NH*HD*NSTATE) return;
    int h = i / (HD*NSTATE);
    float carry = 0.f;
    for (int z=0; z<NCHUNK; z++) {
        prevB[(size_t)z*NH*HD*NSTATE + i] = (__bf16)carry;
        float cdec = __expf(dAcs[(size_t)(z*CHUNKL + CHUNKL-1)*NH + h]);
        carry = carry*cdec + states[(size_t)z*NH*HD*NSTATE + i];
    }
}

// ---------------------------------------------------------------------------
// Yoff: per (z,h): Y[c,p] += sum_n Cr[c,n]*exp(s_c) * prev[h][p][n]
// M=256(c), K=512(n), N=64(p). Wave owns 2 row-tiles x 4 p-tiles.
// ---------------------------------------------------------------------------
__global__ void yoff_kernel(const __bf16* __restrict__ CrB,    // [1024][512]
                            const float* __restrict__ dAcs,
                            const __bf16* __restrict__ prevB,  // [4][32][64][512]
                            float* __restrict__ yacc)
{
    int z = blockIdx.x >> 5, h = blockIdx.x & 31;
    __shared__ float sE[CHUNKL];
    sE[threadIdx.x] = __expf(dAcs[(size_t)(z*CHUNKL + threadIdx.x)*NH + h]);
    __syncthreads();
    int lane = threadIdx.x & 31, wave = threadIdx.x >> 5;
    int half = lane >> 4;
    const __bf16* pv = prevB + (size_t)(z*NH + h)*HD*NSTATE;

    int it0 = 2*wave, it1 = 2*wave + 1;
    int c0 = it0*16 + (lane & 15);
    int c1 = it1*16 + (lane & 15);
    float ec0 = sE[c0], ec1 = sE[c1];
    const __bf16* ar0 = CrB + (size_t)(z*CHUNKL + c0)*NSTATE;
    const __bf16* ar1 = CrB + (size_t)(z*CHUNKL + c1)*NSTATE;

    v8f acc0[4] = {}, acc1[4] = {};
    for (int nb = 0; nb < NSTATE; nb += 32) {
        v16bf b[4];
        #pragma unroll
        for (int pt=0; pt<4; pt++)
            b[pt] = load_b_frag(pv, NSTATE, pt*16 + (lane & 15), nb, lane);
        v16bf a0, a1;
        {
            const __bf16* base = ar0 + nb + 8*half;
            bf16x8 lo = *(const bf16x8*)base;
            bf16x8 hi = *(const bf16x8*)(base + 16);
            #pragma unroll
            for (int e=0; e<8; e++) { a0[e] = (__bf16)((float)lo[e]*ec0); a0[8+e] = (__bf16)((float)hi[e]*ec0); }
        }
        {
            const __bf16* base = ar1 + nb + 8*half;
            bf16x8 lo = *(const bf16x8*)base;
            bf16x8 hi = *(const bf16x8*)(base + 16);
            #pragma unroll
            for (int e=0; e<8; e++) { a1[e] = (__bf16)((float)lo[e]*ec1); a1[8+e] = (__bf16)((float)hi[e]*ec1); }
        }
        #pragma unroll
        for (int pt=0; pt<4; pt++) {
            acc0[pt] = wmma_bf16(a0, b[pt], acc0[pt]);
            acc1[pt] = wmma_bf16(a1, b[pt], acc1[pt]);
        }
    }
    int mr = 8*half;
    #pragma unroll
    for (int pt=0; pt<4; pt++) {
        int p = pt*16 + (lane & 15);
        #pragma unroll
        for (int r=0; r<8; r++) {
            int t0 = z*CHUNKL + it0*16 + mr + r;
            int t1 = z*CHUNKL + it1*16 + mr + r;
            yacc[((size_t)t0*NH + h)*HD + p] += acc0[pt][r];
            yacc[((size_t)t1*NH + h)*HD + p] += acc1[pt][r];
        }
    }
}

// ---------------------------------------------------------------------------
// Combine 4 directions + gated RMSNorm + build bf16 [silu(z0)*x0 , normed] row.
// One block per token t.
// ---------------------------------------------------------------------------
__global__ void combine_norm_kernel(const float* __restrict__ yacc,
                                    const float* __restrict__ skipC,   // [1024][1024]
                                    const float* __restrict__ norm_w,  // [512]
                                    __bf16* __restrict__ catB)         // [1024][768]
{
    int t = blockIdx.x;
    int h = t >> 5, w = t & 31;
    int t_wh = w*32 + h;
    __shared__ float sval[DSSM];
    __shared__ float ssum;
    int tid = threadIdx.x;
    if (tid == 0) ssum = 0.f;
    __syncthreads();
    float local = 0.f;
    for (int d = tid; d < DSSM; d += blockDim.x) {
        int hh = d >> 6, p = d & 63;
        float v = yacc[((size_t)t       *NH +      hh)*HD + p]
                + yacc[((size_t)(1023-t)*NH + 16 + hh)*HD + p]
                + yacc[((size_t)t_wh    *NH +  8 + hh)*HD + p]
                + yacc[((size_t)(1023-t_wh)*NH + 24 + hh)*HD + p];
        float zv = skipC[(size_t)t*L_TOK + DSSM + d];
        float g = v * siluf_(zv);
        sval[d] = g;
        local += g*g;
    }
    atomicAdd(&ssum, local);
    __syncthreads();
    float rms = rsqrtf(ssum / (float)DSSM + 1e-5f);
    for (int d = tid; d < DSSM; d += blockDim.x)
        catB[(size_t)t*DCAT + DMLP + d] = (__bf16)(sval[d] * rms * norm_w[d]);
    for (int d = tid; d < DMLP; d += blockDim.x) {
        float z0 = skipC[(size_t)t*L_TOK + d];
        float x0 = skipC[(size_t)t*L_TOK + DMLP + d];
        catB[(size_t)t*DCAT + d] = (__bf16)(siluf_(z0) * x0);
    }
}

// ---------------------------------------------------------------------------
// Host orchestration
// ---------------------------------------------------------------------------
static inline void gemm(const __bf16* A, const __bf16* Bt, float* C,
                        int M, int N, int Kd, int lda, int ldb, int ldc, hipStream_t s)
{
    int supertiles = (M/32)*(N/32);
    int blocks = (supertiles + 3)/4;
    gemm_bf16_kernel<<<blocks, 128, 0, s>>>(A, Bt, C, M, N, Kd, lda, ldb, ldc);
}

extern "C" void kernel_launch(void* const* d_in, const int* in_sizes, int n_in,
                              void* d_out, int out_size, void* d_ws, size_t ws_size,
                              hipStream_t stream)
{
    const float* u1        = (const float*)d_in[0];
    const float* u2        = (const float*)d_in[1];
    const float* u2c1      = (const float*)d_in[2];
    const float* u1c2      = (const float*)d_in[3];
    const float* W_skip    = (const float*)d_in[4];
    const float* W_xs      = (const float*)d_in[5];
    const float* W_bcdt    = (const float*)d_in[6];
    const float* conv_xs_w = (const float*)d_in[7];
    const float* conv_xs_b = (const float*)d_in[8];
    const float* conv_bc_w = (const float*)d_in[9];
    const float* conv_bc_b = (const float*)d_in[10];
    const float* Ds        = (const float*)d_in[11];
    const float* norm_w    = (const float*)d_in[12];
    const float* W_out     = (const float*)d_in[13];
    const float* dt_bias   = (const float*)d_in[14];
    const float* A_logs    = (const float*)d_in[15];
    float* out = (float*)d_out;

    char* p = (char*)d_ws;
    auto alloc = [&](size_t bytes) -> void* {
        void* r = (void*)p;
        p += (bytes + 255) & ~(size_t)255;
        return r;
    };

    __bf16* wskipT = (__bf16*)alloc((size_t)1024*DMODEL*2);
    __bf16* wxsT   = (__bf16*)alloc((size_t)DSSM*DMODEL*2);
    __bf16* wbcdtT = (__bf16*)alloc((size_t)BCDT_P*DMODEL*2);
    __bf16* woutT  = (__bf16*)alloc((size_t)DMODEL*DCAT*2);
    __bf16* uB[4];
    for (int i=0;i<4;i++) uB[i] = (__bf16*)alloc((size_t)L_TOK*DMODEL*2);
    float*  skipC  = (float*)alloc((size_t)L_TOK*1024*4);
    float*  xsP    = (float*)alloc((size_t)L_TOK*DSSM*4);
    float*  bcP    = (float*)alloc((size_t)L_TOK*BCDT_P*4);
    float*  convXS = (float*)alloc((size_t)DSSM*L_TOK*4);
    float*  convBC = (float*)alloc((size_t)BCDT_C*L_TOK*4);
    float*  dtbuf  = (float*)alloc((size_t)L_TOK*NH*4);
    float*  dAcs   = (float*)alloc((size_t)L_TOK*NH*4);
    __bf16* xdtT   = (__bf16*)alloc((size_t)NH*HD*L_TOK*2);
    __bf16* CrB    = (__bf16*)alloc((size_t)L_TOK*NSTATE*2);
    __bf16* BrB    = (__bf16*)alloc((size_t)L_TOK*NSTATE*2);
    __bf16* BrT    = (__bf16*)alloc((size_t)NSTATE*L_TOK*2);
    float*  CBbuf  = (float*)alloc((size_t)NCHUNK*CHUNKL*CHUNKL*4);
    float*  states = (float*)alloc((size_t)NCHUNK*NH*HD*NSTATE*4);
    __bf16* prevB  = (__bf16*)alloc((size_t)NCHUNK*NH*HD*NSTATE*2);
    float*  yacc   = (float*)alloc((size_t)L_TOK*NH*HD*4);
    __bf16* catB   = (__bf16*)alloc((size_t)L_TOK*DCAT*2);

    // --- one-time weight conversion (N-major bf16 for B-fragments) ---
    {
        int n;
        n = 1024*DMODEL;  transpose_bf16_kernel<<<(n+255)/256,256,0,stream>>>(W_skip, wskipT, DMODEL, 1024, 1024);
        n = DSSM*DMODEL;  transpose_bf16_kernel<<<(n+255)/256,256,0,stream>>>(W_xs,   wxsT,   DMODEL, DSSM, DSSM);
        n = BCDT_P*DMODEL;transpose_bf16_kernel<<<(n+255)/256,256,0,stream>>>(W_bcdt, wbcdtT, DMODEL, BCDT_C, BCDT_P);
        n = DMODEL*DCAT;  transpose_bf16_kernel<<<(n+255)/256,256,0,stream>>>(W_out,  woutT,  DCAT, DMODEL, DMODEL);
        const float* us[4] = {u1, u2, u2c1, u1c2};
        n = L_TOK*DMODEL;
        for (int i=0;i<4;i++)
            f32_to_bf16_kernel<<<(n+255)/256,256,0,stream>>>(us[i], uB[i], n);
    }

    // --- per-stream pipeline (workspace reused sequentially) ---
    for (int s = 0; s < 2; s++) {
        const __bf16* uBs  = (s == 0) ? uB[0] : uB[1];
        const __bf16* ucBs = (s == 0) ? uB[2] : uB[3];

        // input projections
        gemm(uBs,  wskipT, skipC, L_TOK, 1024,   DMODEL, DMODEL, DMODEL, 1024,   stream);
        gemm(uBs,  wxsT,   xsP,   L_TOK, DSSM,   DMODEL, DMODEL, DMODEL, DSSM,   stream);
        gemm(ucBs, wbcdtT, bcP,   L_TOK, BCDT_P, DMODEL, DMODEL, DMODEL, BCDT_P, stream);

        // depthwise conv + silu  (-> channel-major)
        dwconv_silu_kernel<<<(DSSM*L_TOK+255)/256,256,0,stream>>>(xsP, conv_xs_w, conv_xs_b, convXS, DSSM, DSSM);
        dwconv_silu_kernel<<<(BCDT_C*L_TOK+255)/256,256,0,stream>>>(bcP, conv_bc_w, conv_bc_b, convBC, BCDT_C, BCDT_P);

        // direction sequences + dt
        build_seq_kernel<<<(4*L_TOK+255)/256,256,0,stream>>>(convXS, convBC, dt_bias,
                                                             dtbuf, xdtT, CrB, BrB, BrT);
        cumsum_kernel<<<1,NCHUNK*NH,0,stream>>>(dtbuf, A_logs, dAcs);

        // CB = Cr @ Br^T per chunk (shared across heads), K = n = 512
        for (int z = 0; z < NCHUNK; z++)
            gemm(CrB + (size_t)z*CHUNKL*NSTATE, BrB + (size_t)z*CHUNKL*NSTATE,
                 CBbuf + (size_t)z*CHUNKL*CHUNKL,
                 CHUNKL, CHUNKL, NSTATE, NSTATE, NSTATE, CHUNKL, stream);

        // intra-chunk (masked/decayed) contribution + D*x
        ydiag_kernel<<<NCHUNK*NH,CHUNKL,0,stream>>>(CBbuf, dAcs, xdtT, convXS, Ds, yacc);
        // chunk states, inter-chunk recurrence, cross-chunk contribution
        states_kernel<<<NCHUNK*NH,CHUNKL,0,stream>>>(BrT, dAcs, xdtT, states);
        recurrence_kernel<<<(NH*HD*NSTATE+255)/256,256,0,stream>>>(states, dAcs, prevB);
        yoff_kernel<<<NCHUNK*NH,CHUNKL,0,stream>>>(CrB, dAcs, prevB, yacc);

        // direction merge + gated RMSNorm + concat with silu(z0)*x0
        combine_norm_kernel<<<L_TOK,256,0,stream>>>(yacc, skipC, norm_w, catB);

        // output projection straight into d_out
        gemm(catB, woutT, out + (size_t)s*L_TOK*DMODEL,
             L_TOK, DMODEL, DCAT, DCAT, DCAT, DMODEL, stream);
    }
}
